// BurstAlign_36696200577508
// MI455X (gfx1250) — matrix operand order repaired and checked by reading, not compiled, not run on hardware
//
#include <hip/hip_runtime.h>
#include <hip/hip_bf16.h>
#include <math.h>

typedef __attribute__((ext_vector_type(16))) _Float16 v16h;
typedef __attribute__((ext_vector_type(8)))  _Float16 v8h;
typedef __attribute__((ext_vector_type(8)))  float    v8f;

#define HH 160
#define WW 160
#define HW (160*160)
#define NF 64
#define G_ 8
#define KTAP 9

__device__ __forceinline__ int iclamp(int v, int lo, int hi) {
    return v < lo ? lo : (v > hi ? hi : v);
}

// ---------------------------------------------------------------------------
// Weight packing: fp32 source -> f16 WMMA B-fragment layout.
// Fragment layout: wp[((nt*nK + ks)*32 + lane)*16 + h]
//   oc = nt*16 + (lane&15); kk = (h<8) ? 8*(lane>>4) + h : 16 + 8*(lane>>4) + (h-8); kd = ks*32 + kk.
// mode 0: kd = tap*Cin + ic, src = w[oc][ic][tap]   (OIHW conv weights)
// mode 1: src = w[oc*Ktot + kd]                      (deformable-conv weights)
// ---------------------------------------------------------------------------
__global__ void pack_weights(const float* __restrict__ w, _Float16* __restrict__ wp,
                             int Cin, int Cout, int nNt, int nK, int mode) {
    int idx = blockIdx.x * blockDim.x + threadIdx.x;
    int total = nNt * nK * 32 * 16;
    if (idx >= total) return;
    int h    = idx & 15;
    int lane = (idx >> 4) & 31;
    int ks   = (idx >> 9) % nK;
    int nt   = idx / (512 * nK);
    int kk = (h < 8) ? (8 * (lane >> 4) + h) : (16 + 8 * (lane >> 4) + (h - 8));
    int kd = ks * 32 + kk;
    int oc = nt * 16 + (lane & 15);
    float v = 0.f;
    if (oc < Cout) {
        if (mode == 0) {
            int tap = kd / Cin;
            int ic  = kd % Cin;
            v = w[((size_t)oc * Cin + ic) * 9 + tap];
        } else {
            int Ktot = nK * 32;
            v = w[(size_t)oc * Ktot + kd];
        }
    }
    wp[idx] = (_Float16)v;
}

// ---------------------------------------------------------------------------
// First conv: Cin=4 -> 64, direct fp32, relu, emits f16 activations (CHW).
// ---------------------------------------------------------------------------
__global__ __launch_bounds__(256) void conv1_direct(
    const float* __restrict__ x, const float* __restrict__ w,
    const float* __restrict__ b, _Float16* __restrict__ out, int nImg) {
    __shared__ float ws_[NF * 36];
    for (int i = threadIdx.x; i < NF * 36; i += 256) ws_[i] = w[i];
    __syncthreads();
    int t = blockIdx.x * 256 + threadIdx.x;
    if (t >= nImg * HW) return;
    int img = t / HW, p = t % HW;
    int y = p / WW, xx = p % WW;
    const float* xb = x + (size_t)img * 4 * HW;
    float patch[36];
#pragma unroll
    for (int ic = 0; ic < 4; ++ic)
#pragma unroll
        for (int tap = 0; tap < 9; ++tap) {
            int yy = y + tap / 3 - 1, xc = xx + tap % 3 - 1;
            float v = 0.f;
            if (yy >= 0 && yy < HH && xc >= 0 && xc < WW)
                v = xb[(size_t)ic * HW + yy * WW + xc];
            patch[ic * 9 + tap] = v;
        }
    _Float16* ob = out + (size_t)img * NF * HW;
    for (int oc = 0; oc < NF; ++oc) {
        float acc = b[oc];
#pragma unroll
        for (int j = 0; j < 36; ++j) acc = fmaf(patch[j], ws_[oc * 36 + j], acc);
        acc = acc > 0.f ? acc : 0.f;
        ob[(size_t)oc * HW + p] = (_Float16)acc;
    }
}

// ---------------------------------------------------------------------------
// Implicit-GEMM 3x3 conv with WMMA f32_16x16x32_f16.
// One wave per 16-pixel strip (M=16 consecutive x). K = 9*Cin (tap-major).
// Dual-input: ic < CinA from inA (ref feats, strideA may be 0), else inB;
// skipB remaps img -> frame index skipping the center frame.
// LDS patch layout [row 0..2][col 0..17][ic] (ic innermost) so each lane's A
// fragment is two contiguous 16-byte chunks per the ISA 16-bit A layout.
// ---------------------------------------------------------------------------
__global__ __launch_bounds__(32) void conv3x3_wmma(
    const _Float16* __restrict__ inA, long long strideA,
    const _Float16* __restrict__ inB, long long strideB, int skipB,
    const _Float16* __restrict__ wfrag, const float* __restrict__ bias,
    _Float16* __restrict__ out, long long strideOut,
    int CinA, int Cin, int Cout, int nNt, int relu) {
    __shared__ _Float16 patch[3 * 18 * 128];
    const int lane = threadIdx.x;
    const int tile = blockIdx.x;
    const int img  = blockIdx.y;
    const int y  = tile / (WW / 16);
    const int x0 = (tile % (WW / 16)) * 16;

    const _Float16* baseA = inA + (size_t)img * (size_t)strideA;
    const _Float16* baseB = nullptr;
    if (inB) {
        int frameB = img + ((skipB && img >= 2) ? 1 : 0);
        baseB = inB + (size_t)frameB * (size_t)strideB;
    }

    const int tot = 3 * 18 * Cin;
    for (int idx = lane; idx < tot; idx += 32) {
        int ic = idx % Cin;
        int rc = idx / Cin;
        int c = rc % 18, r = rc / 18;
        int yy = y + r - 1, xx = x0 + c - 1;
        _Float16 v = (_Float16)0.f;
        if (yy >= 0 && yy < HH && xx >= 0 && xx < WW) {
            const _Float16* src = (ic < CinA) ? (baseA + (size_t)ic * HW)
                                              : (baseB + (size_t)(ic - CinA) * HW);
            v = src[yy * WW + xx];
        }
        patch[(r * 18 + c) * Cin + ic] = v;
    }
    __syncthreads();

    const int nIcb = Cin / 32;
    const int nK   = nIcb * 9;
    const int m  = lane & 15;
    const int hs = (lane >> 4) & 1;

    for (int nt = 0; nt < nNt; ++nt) {
        v8f acc = {};
        const v16h* wf = (const v16h*)wfrag + (size_t)nt * nK * 32;
        int ks = 0;
        for (int tap = 0; tap < 9; ++tap) {
            int dy = tap / 3, dx = tap % 3;
            int lbase0 = (dy * 18 + (m + dx)) * Cin;
            for (int icb = 0; icb < nIcb; ++icb, ++ks) {
                int lb = lbase0 + icb * 32 + (hs ? 8 : 0);
                v8h a0 = *(const v8h*)(patch + lb);
                v8h a1 = *(const v8h*)(patch + lb + 16);
                v16h a;
#pragma unroll
                for (int i = 0; i < 8; ++i) { a[i] = a0[i]; a[i + 8] = a1[i]; }
                v16h bfrag = wf[(size_t)ks * 32 + lane];
                acc = __builtin_amdgcn_wmma_f32_16x16x32_f16(
                    false, a, false, bfrag, (short)0, acc, false, false);
            }
        }
        int oc = nt * 16 + m;
        if (oc < Cout) {
            float bv = bias ? bias[oc] : 0.f;
            _Float16* po = out + (size_t)img * (size_t)strideOut + (size_t)oc * HW
                           + y * WW + x0 + hs * 8;
            v8h ov;
#pragma unroll
            for (int i = 0; i < 8; ++i) {
                float v = acc[i] + bv;
                if (relu) v = v > 0.f ? v : 0.f;
                ov[i] = (_Float16)v;
            }
            *(v8h*)po = ov;
        }
    }
}

// ---------------------------------------------------------------------------
// MDCN stage 1: bilinear sampling with offsets + sigmoid(mask) modulation.
// Writes samp[pixel][kd], kd = (g*8+c)*9 + k   (K-dim = 576, A-side of GEMM).
// One thread per (pixel, group); the 4 gather addresses/weights are shared
// across the 8 channels of the group.
// ---------------------------------------------------------------------------
__global__ __launch_bounds__(256) void mdcn_sample(
    const _Float16* __restrict__ feats,   // current frame, 64ch CHW
    const _Float16* __restrict__ raw,     // 216ch CHW: 144 offsets + 72 masks
    _Float16* __restrict__ samp) {
    int t = blockIdx.x * 256 + threadIdx.x;
    if (t >= HW * G_) return;
    int p = t % HW, g = t / HW;
    int y = p / WW, x = p % WW;
    for (int k = 0; k < KTAP; ++k) {
        float oy = (float)raw[(size_t)((g * 9 + k) * 2 + 0) * HW + p];
        float ox = (float)raw[(size_t)((g * 9 + k) * 2 + 1) * HW + p];
        float mk = (float)raw[(size_t)(144 + g * 9 + k) * HW + p];
        mk = 1.f / (1.f + expf(-mk));
        float py = oy + (float)(y + k / 3 - 1);
        float px = ox + (float)(x + k % 3 - 1);
        float y0f = floorf(py), x0f = floorf(px);
        float ty = py - y0f, tx = px - x0f;
        int y0 = (int)y0f, x0i = (int)x0f;
        int y1 = y0 + 1, x1 = x0i + 1;
        bool vy0 = (y0 >= 0) & (y0 < HH), vy1 = (y1 >= 0) & (y1 < HH);
        bool vx0 = (x0i >= 0) & (x0i < WW), vx1 = (x1 >= 0) & (x1 < WW);
        int yc0 = iclamp(y0, 0, HH - 1), yc1 = iclamp(y1, 0, HH - 1);
        int xc0 = iclamp(x0i, 0, WW - 1), xc1 = iclamp(x1, 0, WW - 1);
        float w00 = (vy0 && vx0) ? (1.f - ty) * (1.f - tx) * mk : 0.f;
        float w01 = (vy0 && vx1) ? (1.f - ty) * tx * mk : 0.f;
        float w10 = (vy1 && vx0) ? ty * (1.f - tx) * mk : 0.f;
        float w11 = (vy1 && vx1) ? ty * tx * mk : 0.f;
        const _Float16* fb = feats + (size_t)(g * 8) * HW;
#pragma unroll
        for (int c = 0; c < 8; ++c) {
            const _Float16* fc = fb + (size_t)c * HW;
            float v = w00 * (float)fc[yc0 * WW + xc0]
                    + w01 * (float)fc[yc0 * WW + xc1]
                    + w10 * (float)fc[yc1 * WW + xc0]
                    + w11 * (float)fc[yc1 * WW + xc1];
            samp[(size_t)p * 576 + (size_t)((g * 8 + c) * 9 + k)] = (_Float16)v;
        }
    }
}

// ---------------------------------------------------------------------------
// MDCN stage 2: GEMM M=25600(pixels) x K=576 x N=64 via WMMA, fp32 out.
// ---------------------------------------------------------------------------
__global__ __launch_bounds__(32) void mdcn_gemm(
    const _Float16* __restrict__ samp, const _Float16* __restrict__ wfrag,
    const float* __restrict__ bias, float* __restrict__ outFrame) {
    const int lane = threadIdx.x;
    const int pix0 = blockIdx.x * 16;
    const int m  = lane & 15;
    const int hs = (lane >> 4) & 1;
    const _Float16* arow = samp + (size_t)(pix0 + m) * 576 + (hs ? 8 : 0);
    for (int nt = 0; nt < 4; ++nt) {
        v8f acc = {};
        const v16h* wf = (const v16h*)wfrag + (size_t)nt * 18 * 32;
        for (int ks = 0; ks < 18; ++ks) {
            const _Float16* ap = arow + ks * 32;
            v8h a0 = *(const v8h*)ap;
            v8h a1 = *(const v8h*)(ap + 16);
            v16h a;
#pragma unroll
            for (int i = 0; i < 8; ++i) { a[i] = a0[i]; a[i + 8] = a1[i]; }
            v16h bfrag = wf[(size_t)ks * 32 + lane];
            acc = __builtin_amdgcn_wmma_f32_16x16x32_f16(
                false, a, false, bfrag, (short)0, acc, false, false);
        }
        int oc = nt * 16 + m;
        float bv = bias[oc];
        float* po = outFrame + (size_t)oc * HW + pix0 + hs * 8;
#pragma unroll
        for (int i = 0; i < 8; ++i) po[i] = acc[i] + bv;
    }
}

// Copy reference frame features (f16) into output slot 2 (fp32).
__global__ __launch_bounds__(256) void copy_ref(const _Float16* __restrict__ f,
                                                float* __restrict__ o, int n) {
    int t = blockIdx.x * 256 + threadIdx.x;
    if (t < n) o[t] = (float)f[t];
}

// ---------------------------------------------------------------------------
extern "C" void kernel_launch(void* const* d_in, const int* in_sizes, int n_in,
                              void* d_out, int out_size, void* d_ws, size_t ws_size,
                              hipStream_t stream) {
    const float* x   = (const float*)d_in[0];
    const float* fw1 = (const float*)d_in[1];  const float* fb1 = (const float*)d_in[2];
    const float* fw2 = (const float*)d_in[3];  const float* fb2 = (const float*)d_in[4];
    const float* fw3 = (const float*)d_in[5];  const float* fb3 = (const float*)d_in[6];
    const float* ow1 = (const float*)d_in[7];  const float* ob1 = (const float*)d_in[8];
    const float* ow2 = (const float*)d_in[9];  const float* ob2 = (const float*)d_in[10];
    const float* ow3 = (const float*)d_in[11]; const float* ob3 = (const float*)d_in[12];
    const float* dw  = (const float*)d_in[13]; const float* db  = (const float*)d_in[14];
    float* out = (float*)d_out;

    // workspace carving (bytes, 256-aligned)
    size_t cur = 0;
    auto alloc = [&](size_t bytes) {
        size_t o = cur;
        cur = (cur + bytes + 255) & ~(size_t)255;
        return o;
    };
    char* ws = (char*)d_ws;
    const size_t H2 = sizeof(_Float16);
    size_t off_wpF2 = alloc((size_t)4  * 18 * 512 * H2);
    size_t off_wpF3 = alloc((size_t)4  * 18 * 512 * H2);
    size_t off_wpO1 = alloc((size_t)4  * 36 * 512 * H2);
    size_t off_wpO2 = alloc((size_t)4  * 18 * 512 * H2);
    size_t off_wpO3 = alloc((size_t)14 * 18 * 512 * H2);
    size_t off_wpD  = alloc((size_t)4  * 18 * 512 * H2);
    size_t off_fA   = alloc((size_t)5 * NF * HW * H2);
    size_t off_fB   = alloc((size_t)5 * NF * HW * H2);
    size_t off_o1   = alloc((size_t)4 * NF * HW * H2);
    size_t off_o2   = alloc((size_t)4 * NF * HW * H2);
    size_t off_raw  = alloc((size_t)4 * 216 * HW * H2);
    size_t off_samp = alloc((size_t)HW * 576 * H2);

    _Float16* wpF2 = (_Float16*)(ws + off_wpF2);
    _Float16* wpF3 = (_Float16*)(ws + off_wpF3);
    _Float16* wpO1 = (_Float16*)(ws + off_wpO1);
    _Float16* wpO2 = (_Float16*)(ws + off_wpO2);
    _Float16* wpO3 = (_Float16*)(ws + off_wpO3);
    _Float16* wpD  = (_Float16*)(ws + off_wpD);
    _Float16* fA   = (_Float16*)(ws + off_fA);
    _Float16* fB   = (_Float16*)(ws + off_fB);
    _Float16* o1   = (_Float16*)(ws + off_o1);
    _Float16* o2   = (_Float16*)(ws + off_o2);
    _Float16* raw  = (_Float16*)(ws + off_raw);
    _Float16* samp = (_Float16*)(ws + off_samp);

    auto packGrid = [](int nNt, int nK) { return dim3((nNt * nK * 512 + 255) / 256); };
    pack_weights<<<packGrid(4, 18),  256, 0, stream>>>(fw2, wpF2,  64,  64,  4, 18, 0);
    pack_weights<<<packGrid(4, 18),  256, 0, stream>>>(fw3, wpF3,  64,  64,  4, 18, 0);
    pack_weights<<<packGrid(4, 36),  256, 0, stream>>>(ow1, wpO1, 128,  64,  4, 36, 0);
    pack_weights<<<packGrid(4, 18),  256, 0, stream>>>(ow2, wpO2,  64,  64,  4, 18, 0);
    pack_weights<<<packGrid(14, 18), 256, 0, stream>>>(ow3, wpO3,  64, 216, 14, 18, 0);
    pack_weights<<<packGrid(4, 18),  256, 0, stream>>>(dw,  wpD,    0,  64,  4, 18, 1);

    // feature extraction
    conv1_direct<<<(5 * HW + 255) / 256, 256, 0, stream>>>(x, fw1, fb1, fA, 5);

    const long long sF = (long long)NF * HW;
    dim3 g5(HW / 16, 5), g4(HW / 16, 4);
    conv3x3_wmma<<<g5, 32, 0, stream>>>(fA, sF, nullptr, 0, 0, wpF2, fb2, fB, sF,
                                        64, 64, 64, 4, 1);
    conv3x3_wmma<<<g5, 32, 0, stream>>>(fB, sF, nullptr, 0, 0, wpF3, fb3, fA, sF,
                                        64, 64, 64, 4, 1);
    // offset branch: cat(ref, curr) consumed in-place via dual-input loader
    conv3x3_wmma<<<g4, 32, 0, stream>>>(fA + (size_t)2 * NF * HW, 0,   // ref (fixed)
                                        fA, sF, 1,                     // curr (skip center)
                                        wpO1, ob1, o1, sF, 64, 128, 64, 4, 1);
    conv3x3_wmma<<<g4, 32, 0, stream>>>(o1, sF, nullptr, 0, 0, wpO2, ob2, o2, sF,
                                        64, 64, 64, 4, 1);
    conv3x3_wmma<<<g4, 32, 0, stream>>>(o2, sF, nullptr, 0, 0, wpO3, ob3, raw,
                                        (long long)216 * HW, 64, 64, 216, 14, 0);

    // deformable alignment, one frame at a time (samp buffer reused; stream-ordered)
    for (int f = 0; f < 4; ++f) {
        int frame = f + (f >= 2 ? 1 : 0);
        const _Float16* curr = fA + (size_t)frame * NF * HW;
        const _Float16* rawF = raw + (size_t)f * 216 * HW;
        mdcn_sample<<<(HW * G_ + 255) / 256, 256, 0, stream>>>(curr, rawF, samp);
        mdcn_gemm<<<HW / 16, 32, 0, stream>>>(samp, wpD, db,
                                              out + (size_t)frame * NF * HW);
    }
    // center frame passthrough
    copy_ref<<<(NF * HW + 255) / 256, 256, 0, stream>>>(
        fA + (size_t)2 * NF * HW, out + (size_t)2 * NF * HW, NF * HW);
}
